// VSA_8976481648867
// MI455X (gfx1250) — compile-verified, matrix operand order checked by reading, not run on hardware
//
#include <hip/hip_runtime.h>

// ---------------------------------------------------------------------------
// VSA bottom-up tree encode as circulant-matrix GEMMs on CDNA5 WMMA
// (v_wmma_f32_16x16x32_bf16, wave32).
//
//   circ(role, x)[i] = sum_k x[k] * role[(i-k) mod D]   ==>   y = x * W,
//   W[k][n] = role[(n-k) mod D]  (stored transposed, N-major, for B frags)
//
// Level recurrence (flattened rows p = b*2^d + j; children are rows 2p, 2p+1):
//   rep[p] = valid[p] * ( filler[t[p]-1] + child[2p]*W_L + child[2p+1]*W_R )
// ---------------------------------------------------------------------------

#define DIM     1024
#define DEPTH   8
#define BATCH   256
#define NODES   255   // 2^DEPTH - 1

typedef __bf16 bf16_t;
typedef __attribute__((ext_vector_type(16))) __bf16 v16bf;
typedef __attribute__((ext_vector_type(8)))  __bf16 v8bf;
typedef __attribute__((ext_vector_type(4)))  __bf16 v4bf;
typedef __attribute__((ext_vector_type(8)))  float  v8f;

union FragAB { v16bf f; v8bf h[2]; };

// ---------------------------------------------------------------------------
// Circulant role matrices, transposed (row n holds W[:,n] over K), bf16.
//   WtX[n*DIM + k] = role[(n - k) & (DIM-1)]
// WtR is REQUIRED to sit exactly DIM*DIM elements after WtL (single-base
// B-fragment addressing in level_kernel relies on it).
// ---------------------------------------------------------------------------
__global__ __launch_bounds__(256) void circ_kernel(
    const float* __restrict__ roles, bf16_t* __restrict__ WtL,
    bf16_t* __restrict__ WtR)
{
    int gid = blockIdx.x * blockDim.x + threadIdx.x;   // DIM*DIM threads
    int n   = gid >> 10;
    int k   = gid & (DIM - 1);
    int idx = (n - k) & (DIM - 1);
    WtL[gid] = (bf16_t)roles[idx];
    WtR[gid] = (bf16_t)roles[DIM + idx];
}

// ---------------------------------------------------------------------------
// Leaf level (d = DEPTH-1): rep = valid ? filler[t-1] : 0, stored bf16.
// Rows = BATCH * 2^(DEPTH-1) = 32768, cols = DIM.  4 elems/thread.
// ---------------------------------------------------------------------------
__global__ __launch_bounds__(256) void leaf_kernel(
    const int* __restrict__ trees, const float* __restrict__ filler,
    bf16_t* __restrict__ out)
{
    int gid = blockIdx.x * blockDim.x + threadIdx.x;
    int row = gid >> 8;              // DIM/4 = 256 threads per row
    int c4  = (gid & 255) << 2;
    int b   = row >> (DEPTH - 1);
    int j   = row & ((1 << (DEPTH - 1)) - 1);
    int t   = trees[b * NODES + ((1 << (DEPTH - 1)) - 1) + j];

    bf16_t z = (bf16_t)0.0f;
    v4bf o = {z, z, z, z};
    if (t > 0) {
        const float4 f = *(const float4*)(filler + (size_t)(t - 1) * DIM + c4);
        o[0] = (bf16_t)f.x; o[1] = (bf16_t)f.y;
        o[2] = (bf16_t)f.z; o[3] = (bf16_t)f.w;
    }
    *(v4bf*)(out + (size_t)row * DIM + c4) = o;
}

// ---------------------------------------------------------------------------
// Internal level: one wave computes a 16-row x 64-col tile.
// All 20 b128 loads of a K-step are issued (off two stepped base pointers,
// immediate offsets) before the 8 chained WMMAs, so the scheduler can use
// partial s_wait_loadcnt and overlap WMMA with tail loads.
// ---------------------------------------------------------------------------
__global__ __launch_bounds__(256) void level_kernel(
    const int*    __restrict__ trees, const float* __restrict__ filler,
    const bf16_t* __restrict__ WtL,   /* WtR = WtL + DIM*DIM */
    const bf16_t* __restrict__ child, bf16_t* __restrict__ outB,
    float*        __restrict__ outF,  int nshift)
{
    const int lane  = threadIdx.x & 31;
    const int wid   = blockIdx.x * (blockDim.x >> 5) + (threadIdx.x >> 5);
    const int colTiles = DIM / 64;                  // 16 column groups
    const int rowBase  = (wid / colTiles) * 16;
    const int colBase  = (wid % colTiles) * 64;

    const int laneM = lane & 15;
    const int kHalf = (lane < 16) ? 0 : 8;          // A/B 16-bit VGPR layout

    // A base: lane laneM supplies matrix row M=laneM -> parent p.
    // Left child = row 2p (offset 0), right child = row 2p+1 (offset +DIM).
    const int p = rowBase + laneM;
    const bf16_t* aB = child + (size_t)(2 * p) * DIM + kHalf;

    // B base: lane laneM supplies column n = colBase + laneM of W_L.
    // ct-tile at +ct*16*DIM, W_R fragments at +DIM*DIM (2 MB, within ioffset).
    const bf16_t* bB = WtL + (size_t)(colBase + laneM) * DIM + kHalf;

    v8f acc[4] = {};

#pragma unroll 1
    for (int k = 0; k < DIM; k += 32) {
        FragAB AL, AR, BL[4], BR[4];
        // ---- issue every load of this K-step first ----
        AL.h[0] = *(const v8bf*)(aB);
        AL.h[1] = *(const v8bf*)(aB + 16);
        AR.h[0] = *(const v8bf*)(aB + DIM);
        AR.h[1] = *(const v8bf*)(aB + DIM + 16);
#pragma unroll
        for (int ct = 0; ct < 4; ++ct) {
            const bf16_t* pL = bB + (size_t)ct * 16 * DIM;
            BL[ct].h[0] = *(const v8bf*)(pL);
            BL[ct].h[1] = *(const v8bf*)(pL + 16);
            BR[ct].h[0] = *(const v8bf*)(pL + (size_t)DIM * DIM);
            BR[ct].h[1] = *(const v8bf*)(pL + (size_t)DIM * DIM + 16);
        }
        // ---- then the 8 chained WMMAs ----
#pragma unroll
        for (int ct = 0; ct < 4; ++ct) {
            acc[ct] = __builtin_amdgcn_wmma_f32_16x16x32_bf16(
                false, AL.f, false, BL[ct].f, (short)0, acc[ct], false, false);
            acc[ct] = __builtin_amdgcn_wmma_f32_16x16x32_bf16(
                false, AR.f, false, BR[ct].f, (short)0, acc[ct], false, false);
        }
        aB += 32;
        bB += 32;
    }

    // Epilogue: C/D layout -> VGPR r: M = r + (lane<16 ? 0 : 8), N = lane&15.
    const int mOff  = (lane < 16) ? 0 : 8;
    const int nNode = 1 << nshift;
    const int start = nNode - 1;
    if (outF != nullptr) {
#pragma unroll
        for (int ct = 0; ct < 4; ++ct) {
            int col = colBase + ct * 16 + laneM;
#pragma unroll
            for (int r = 0; r < 8; ++r) {
                int row = rowBase + r + mOff;
                int b   = row >> nshift;
                int j   = row & (nNode - 1);
                int t   = trees[b * NODES + start + j];
                float v = 0.0f;
                if (t > 0)
                    v = acc[ct][r] + filler[(size_t)(t - 1) * DIM + col];
                outF[(size_t)row * DIM + col] = v;
            }
        }
    } else {
#pragma unroll
        for (int ct = 0; ct < 4; ++ct) {
            int col = colBase + ct * 16 + laneM;
#pragma unroll
            for (int r = 0; r < 8; ++r) {
                int row = rowBase + r + mOff;
                int b   = row >> nshift;
                int j   = row & (nNode - 1);
                int t   = trees[b * NODES + start + j];
                float v = 0.0f;
                if (t > 0)
                    v = acc[ct][r] + filler[(size_t)(t - 1) * DIM + col];
                outB[(size_t)row * DIM + col] = (bf16_t)v;
            }
        }
    }
}

// ---------------------------------------------------------------------------
// Host-side orchestration (graph-capture safe: only kernel launches).
// Workspace layout (bf16):
//   [0, 2MB)    WtL        1024x1024
//   [2, 4MB)    WtR        1024x1024   (must be contiguous after WtL)
//   [4, 68MB)   buf0       32768 x 1024   (levels 7,5,3,1)
//   [68,100MB)  buf1       16384 x 1024   (levels 6,4,2)
// ---------------------------------------------------------------------------
extern "C" void kernel_launch(void* const* d_in, const int* in_sizes, int n_in,
                              void* d_out, int out_size, void* d_ws, size_t ws_size,
                              hipStream_t stream) {
    const int*   trees  = (const int*)d_in[0];
    const float* filler = (const float*)d_in[1];
    const float* roles  = (const float*)d_in[2];
    float*       out    = (float*)d_out;

    bf16_t* WtL  = (bf16_t*)d_ws;
    bf16_t* WtR  = WtL  + (size_t)DIM * DIM;
    bf16_t* buf0 = WtR  + (size_t)DIM * DIM;
    bf16_t* buf1 = buf0 + (size_t)(BATCH << (DEPTH - 1)) * DIM;

    // 1) circulant role matrices (bf16, transposed)
    circ_kernel<<<(DIM * DIM) / 256, 256, 0, stream>>>(roles, WtL, WtR);

    // 2) leaf level d = 7
    {
        size_t elems = (size_t)(BATCH << (DEPTH - 1)) * DIM;
        leaf_kernel<<<(unsigned)(elems / 4 / 256), 256, 0, stream>>>(trees, filler, buf0);
    }

    // 3) levels d = 6 .. 0
    bf16_t* cur = buf0;
    bf16_t* nxt = buf1;
    for (int d = DEPTH - 2; d >= 0; --d) {
        int mrows  = BATCH << d;
        int tiles  = (mrows / 16) * (DIM / 64);
        int blocks = tiles / 8;                  // 8 waves per 256-thread block
        level_kernel<<<blocks, 256, 0, stream>>>(
            trees, filler, WtL, cur,
            (d == 0) ? nullptr : nxt,
            (d == 0) ? out : nullptr,
            d);
        bf16_t* tmp = cur; cur = nxt; nxt = tmp;
    }
}
